// GridExtractor_49443663511572
// MI455X (gfx1250) — compile-verified, matrix operand order checked by reading, not run on hardware
//
#include <hip/hip_runtime.h>
#include <hip/hip_bf16.h>

typedef unsigned short ushort_t;
typedef __attribute__((ext_vector_type(16))) __bf16 v16bf;
typedef __attribute__((ext_vector_type(8)))  float  v8f;
typedef __attribute__((ext_vector_type(4))) unsigned u32x4;
typedef __attribute__((ext_vector_type(8))) int      i32x8;
typedef __attribute__((ext_vector_type(4))) int      i32x4;

#define B_  4
#define N_  1024
#define CIN_ 256
#define H_  256
#define W_  256
#define D_  512
#define IN_ 2304          // CIN * 3 * 3
#define ROWS_ 4096        // B * N
#define SCALE_ 0.25f
#define INV_IMG_ (1.0f / 1024.0f)

__device__ __forceinline__ ushort_t f2bf(float f) {
  unsigned u = __float_as_uint(f);
  unsigned r = (u + 0x7FFFu + ((u >> 16) & 1u)) >> 16;   // round-to-nearest-even
  return (ushort_t)r;
}

// ---------------------------------------------------------------------------
// TDM: DMA a 2-D bf16 tile (tile_rows x 32 halves, row stride = stride_elems)
// from global memory into LDS at lds_off.  TDM pad: after every 16 DWORDs
// (one 32-half row) skip 4 DWORDs -> LDS row pitch = 40 halves.
// ---------------------------------------------------------------------------
__device__ __forceinline__ void tdm_load_tile(unsigned lds_off,
                                              const void* gptr,
                                              int tile_rows,
                                              unsigned long long stride_elems) {
  unsigned long long ga = (unsigned long long)(uintptr_t)gptr;
  u32x4 g0;
  g0[0] = 1u;                                            // count=1, user mode
  g0[1] = lds_off;                                       // lds_addr (bytes)
  g0[2] = (unsigned)ga;                                  // global_addr[31:0]
  g0[3] = ((unsigned)(ga >> 32) & 0x01FFFFFFu) | (2u << 30); // [56:32] | type=2
  i32x8 g1;
  g1[0] = (int)((1u << 16) |          // data_size = 2 bytes
                (1u << 20) |          // pad_enable
                (3u << 22) |          // pad_interval: 16 DWORDs (one row)
                (3u << 25));          // pad_amount:   4 DWORDs
  g1[1] = (int)(32u << 16);                              // tensor_dim0 = 32
  g1[2] = (int)(((unsigned)tile_rows) << 16);            // tensor_dim1 = rows
  g1[3] = (int)(32u << 16);                              // tile_dim0 = 32
  g1[4] = tile_rows;                                     // tile_dim1 (dim2=0)
  g1[5] = (int)(unsigned)stride_elems;                   // dim0_stride lo32
  g1[6] = (int)(unsigned)(stride_elems >> 32);           // dim0_stride hi16
  g1[7] = 0;
  i32x4 z4 = {0, 0, 0, 0};
#if defined(__clang_major__) && (__clang_major__ >= 23)
  i32x8 z8 = {0, 0, 0, 0, 0, 0, 0, 0};
  __builtin_amdgcn_tensor_load_to_lds(g0, g1, z4, z4, z8, 0);
#else
  __builtin_amdgcn_tensor_load_to_lds(g0, g1, z4, z4, 0);
#endif
}

// ---------------------------------------------------------------------------
// Kernel 1: bbox min/max from 8-pt polygon, positional GEMM (K=4) + LayerNorm1
// ---------------------------------------------------------------------------
__global__ __launch_bounds__(256) void bbox_pos_kernel(
    const float* __restrict__ boxes, const float* __restrict__ Wb,
    const float* __restrict__ bb, const float* __restrict__ g1,
    const float* __restrict__ be1, float* __restrict__ bboxOut,
    float* __restrict__ pos) {
  const int row = blockIdx.x;
  const int t = threadIdx.x;
  const float* bx = boxes + (size_t)row * 8;
  float x0 = bx[0], y0 = bx[1], x1 = bx[2], y1 = bx[3];
  float x2 = bx[4], y2 = bx[5], x3 = bx[6], y3 = bx[7];
  float X1 = fminf(fminf(x0, x1), fminf(x2, x3));
  float Y1 = fminf(fminf(y0, y1), fminf(y2, y3));
  float X2 = fmaxf(fmaxf(x0, x1), fmaxf(x2, x3));
  float Y2 = fmaxf(fmaxf(y0, y1), fmaxf(y2, y3));
  if (t == 0) {
    bboxOut[row * 4 + 0] = X1; bboxOut[row * 4 + 1] = Y1;
    bboxOut[row * 4 + 2] = X2; bboxOut[row * 4 + 3] = Y2;
  }
  const float nb0 = X1 * INV_IMG_, nb1 = Y1 * INV_IMG_;
  const float nb2 = X2 * INV_IMG_, nb3 = Y2 * INV_IMG_;

  float raw[2];
#pragma unroll
  for (int i = 0; i < 2; ++i) {
    int d = t + i * 256;
    raw[i] = nb0 * Wb[d] + nb1 * Wb[D_ + d] + nb2 * Wb[2 * D_ + d] +
             nb3 * Wb[3 * D_ + d] + bb[d];
  }

  __shared__ float red[256];
  red[t] = raw[0] + raw[1];
  __syncthreads();
  for (int o = 128; o > 0; o >>= 1) {
    if (t < o) red[t] += red[t + o];
    __syncthreads();
  }
  float mean = red[0] * (1.0f / D_);
  __syncthreads();
  float d0 = raw[0] - mean, d1 = raw[1] - mean;
  red[t] = d0 * d0 + d1 * d1;
  __syncthreads();
  for (int o = 128; o > 0; o >>= 1) {
    if (t < o) red[t] += red[t + o];
    __syncthreads();
  }
  float inv_std = rsqrtf(red[0] * (1.0f / D_) + 1e-5f);
#pragma unroll
  for (int i = 0; i < 2; ++i) {
    int d = t + i * 256;
    pos[(size_t)row * D_ + d] = (raw[i] - mean) * inv_std * g1[d] + be1[d];
  }
}

// ---------------------------------------------------------------------------
// Kernel 2: RoI align.  One block per box, one lane per channel; bf16 output.
// ---------------------------------------------------------------------------
__global__ __launch_bounds__(256) void roi_kernel(
    const float* __restrict__ feats, const float* __restrict__ bbox,
    ushort_t* __restrict__ flat) {
  const int box = blockIdx.x;
  const int b = box >> 10;
  const int c = threadIdx.x;
  const float X1 = bbox[box * 4 + 0] * SCALE_;
  const float Y1 = bbox[box * 4 + 1] * SCALE_;
  const float X2 = bbox[box * 4 + 2] * SCALE_;
  const float Y2 = bbox[box * 4 + 3] * SCALE_;
  const float rw = fmaxf(X2 - X1, 1.0f);
  const float rh = fmaxf(Y2 - Y1, 1.0f);
  const float binw = rw * (1.0f / 3.0f);
  const float binh = rh * (1.0f / 3.0f);
  const float* __restrict__ img = feats + ((size_t)(b * CIN_ + c)) * (H_ * W_);
  ushort_t* __restrict__ dst = flat + (size_t)box * IN_ + c * 9;

#pragma unroll
  for (int ph = 0; ph < 3; ++ph) {
#pragma unroll
    for (int pw = 0; pw < 3; ++pw) {
      float acc = 0.0f;
#pragma unroll
      for (int iy = 0; iy < 2; ++iy) {
#pragma unroll
        for (int ix = 0; ix < 2; ++ix) {
          float yy = Y1 + ((float)ph + ((float)iy + 0.5f) * 0.5f) * binh;
          float xx = X1 + ((float)pw + ((float)ix + 0.5f) * 0.5f) * binw;
          bool valid = (yy > -1.0f) && (yy < (float)H_) &&
                       (xx > -1.0f) && (xx < (float)W_);
          float y = fminf(fmaxf(yy, 0.0f), (float)(H_ - 1));
          float x = fminf(fmaxf(xx, 0.0f), (float)(W_ - 1));
          int yi0 = (int)floorf(y);
          int xi0 = (int)floorf(x);
          int yi1 = min(yi0 + 1, H_ - 1);
          int xi1 = min(xi0 + 1, W_ - 1);
          float ly = y - (float)yi0, lx = x - (float)xi0;
          float hy = 1.0f - ly, hx = 1.0f - lx;
          float v = img[yi0 * W_ + xi0] * (hy * hx) +
                    img[yi0 * W_ + xi1] * (hy * lx) +
                    img[yi1 * W_ + xi0] * (ly * hx) +
                    img[yi1 * W_ + xi1] * (ly * lx);
          acc += valid ? v : 0.0f;
        }
      }
      dst[ph * 3 + pw] = f2bf(acc * 0.25f);
    }
  }
}

// ---------------------------------------------------------------------------
// Kernel 3: f32 (K x 512) -> bf16 transposed (512 x K).  32x32 LDS tiles.
// ---------------------------------------------------------------------------
__global__ __launch_bounds__(256) void cvt_tr_kernel(
    const float* __restrict__ s, ushort_t* __restrict__ d, int K) {
  __shared__ float tile[32][33];
  const int kt = blockIdx.x * 32;
  const int nt = blockIdx.y * 32;
  const int tx = threadIdx.x;   // 0..31
  const int ty = threadIdx.y;   // 0..7
#pragma unroll
  for (int i = 0; i < 4; ++i)
    tile[ty + i * 8][tx] = s[(size_t)(kt + ty + i * 8) * D_ + nt + tx];
  __syncthreads();
#pragma unroll
  for (int i = 0; i < 4; ++i)
    d[(size_t)(nt + ty + i * 8) * K + kt + tx] = f2bf(tile[tx][ty + i * 8]);
}

// ---------------------------------------------------------------------------
// Kernel 4: bf16 WMMA GEMM with TDM double-buffered tile staging.
//   C(4096 x 512) = act(A(4096 x K) * Bt(512 x K)^T + bias)
// Block: 256 threads (8 waves), block tile 128(M) x 128(N); 8 waves as
// 4(M) x 2(N), each wave 32x64 = 2x4 v_wmma_f32_16x16x32_bf16 (8 WMMAs fed
// by 12 ds_load_b128 -> 1.5 LDS b128 reads per WMMA).
// Wave 0 issues both tile DMAs (TENSOR_LOAD_TO_LDS is in-order per wave on
// TENSORcnt), ping-pong LDS buffers, row pitch 40 halves via TDM padding.
// Fragment layout (ISA 16-bit operand): lane L holds row/col (L&15);
// 32-bit half-pair v covers K = (v>=4?16:0) + (L>=16?8:0) + (v&3)*2.
// ---------------------------------------------------------------------------
template <int K, bool RELU, bool OUTBF16>
__global__ __launch_bounds__(256) void gemm_kernel(
    const ushort_t* __restrict__ A, const ushort_t* __restrict__ Bt,
    const float* __restrict__ bias, void* __restrict__ Cout) {
  __shared__ ushort_t As[2][128 * 40];
  __shared__ ushort_t Bs[2][128 * 40];

  const int tid = threadIdx.x;
  const int lane = tid & 31;
  const int wid = tid >> 5;
  const int waveM = wid & 3;          // 4 * 32 = 128 rows
  const int waveN = wid >> 2;         // 2 * 64 = 128 cols
  const int blockN = blockIdx.x * 128;
  const int blockM = blockIdx.y * 128;

  const unsigned asOff[2] = {(unsigned)(uintptr_t)(void*)&As[0][0],
                             (unsigned)(uintptr_t)(void*)&As[1][0]};
  const unsigned bsOff[2] = {(unsigned)(uintptr_t)(void*)&Bs[0][0],
                             (unsigned)(uintptr_t)(void*)&Bs[1][0]};

  const ushort_t* aPtr = A + (size_t)blockM * K;   // advances by 32 elems/iter
  const ushort_t* bPtr = Bt + (size_t)blockN * K;

  // prologue: DMA tiles for k0 = 0 into buffer 0
  if (wid == 0) {
    tdm_load_tile(asOff[0], aPtr, 128, (unsigned long long)K);
    tdm_load_tile(bsOff[0], bPtr, 128, (unsigned long long)K);
    __builtin_amdgcn_s_wait_tensorcnt((short)0);
  }
  __syncthreads();

  const v8f zf = {0.f, 0.f, 0.f, 0.f, 0.f, 0.f, 0.f, 0.f};
  v8f acc[2][4] = {{zf, zf, zf, zf}, {zf, zf, zf, zf}};

  for (int k0 = 0; k0 < K; k0 += 32) {
    const int cur = (k0 >> 5) & 1;
    const int nxt = cur ^ 1;
    // kick off DMA for the next K-chunk while computing on the current one
    if (wid == 0 && (k0 + 32 < K)) {
      tdm_load_tile(asOff[nxt], aPtr + (k0 + 32), 128, (unsigned long long)K);
      tdm_load_tile(bsOff[nxt], bPtr + (k0 + 32), 128, (unsigned long long)K);
    }

    const unsigned* As32 = (const unsigned*)&As[cur][0];
    const unsigned* Bs32 = (const unsigned*)&Bs[cur][0];
    const int hi = (lane >= 16) ? 8 : 0;
    union FragU { unsigned u[8]; v16bf v; };
    FragU a[2], bfr[4];
#pragma unroll
    for (int s = 0; s < 2; ++s) {
      int m = waveM * 32 + s * 16 + (lane & 15);
#pragma unroll
      for (int v = 0; v < 8; ++v) {
        int Kbase = ((v >= 4) ? 16 : 0) + hi + (v & 3) * 2;
        a[s].u[v] = As32[m * 20 + (Kbase >> 1)];
      }
    }
#pragma unroll
    for (int j = 0; j < 4; ++j) {
      int n = waveN * 64 + j * 16 + (lane & 15);
#pragma unroll
      for (int v = 0; v < 8; ++v) {
        int Kbase = ((v >= 4) ? 16 : 0) + hi + (v & 3) * 2;
        bfr[j].u[v] = Bs32[n * 20 + (Kbase >> 1)];
      }
    }
#pragma unroll
    for (int i = 0; i < 2; ++i)
#pragma unroll
      for (int j = 0; j < 4; ++j)
        acc[i][j] = __builtin_amdgcn_wmma_f32_16x16x32_bf16(
            false, a[i].v, false, bfr[j].v, (short)0, acc[i][j], false, false);

    if (wid == 0) __builtin_amdgcn_s_wait_tensorcnt((short)0);
    __syncthreads();
  }

  // epilogue: bias (+ReLU); C/D layout: lanes 0-15 M=vg, lanes 16-31 M=vg+8
#pragma unroll
  for (int i = 0; i < 2; ++i) {
#pragma unroll
    for (int j = 0; j < 4; ++j) {
      int gn = blockN + waveN * 64 + j * 16 + (lane & 15);
      float bv = bias[gn];
#pragma unroll
      for (int vg = 0; vg < 8; ++vg) {
        int gm = blockM + waveM * 32 + i * 16 + ((lane >= 16) ? 8 : 0) + vg;
        float val = acc[i][j][vg] + bv;
        if (RELU) val = fmaxf(val, 0.0f);
        if (OUTBF16)
          ((ushort_t*)Cout)[(size_t)gm * D_ + gn] = f2bf(val);
        else
          ((float*)Cout)[(size_t)gm * D_ + gn] = val;
      }
    }
  }
}

// ---------------------------------------------------------------------------
// Kernel 5: out = LayerNorm2(lines + pos), row-major (4096 x 512)
// ---------------------------------------------------------------------------
__global__ __launch_bounds__(256) void ln2_kernel(
    const float* __restrict__ lines, const float* __restrict__ pos,
    const float* __restrict__ g2, const float* __restrict__ be2,
    float* __restrict__ outrm) {
  const int row = blockIdx.x;
  const int t = threadIdx.x;
  float val[2];
#pragma unroll
  for (int i = 0; i < 2; ++i) {
    int d = t + i * 256;
    val[i] = lines[(size_t)row * D_ + d] + pos[(size_t)row * D_ + d];
  }
  __shared__ float red[256];
  red[t] = val[0] + val[1];
  __syncthreads();
  for (int o = 128; o > 0; o >>= 1) {
    if (t < o) red[t] += red[t + o];
    __syncthreads();
  }
  float mean = red[0] * (1.0f / D_);
  __syncthreads();
  float d0 = val[0] - mean, d1 = val[1] - mean;
  red[t] = d0 * d0 + d1 * d1;
  __syncthreads();
  for (int o = 128; o > 0; o >>= 1) {
    if (t < o) red[t] += red[t + o];
    __syncthreads();
  }
  float inv_std = rsqrtf(red[0] * (1.0f / D_) + 1e-5f);
#pragma unroll
  for (int i = 0; i < 2; ++i) {
    int d = t + i * 256;
    outrm[(size_t)row * D_ + d] = (val[i] - mean) * inv_std * g2[d] + be2[d];
  }
}

// ---------------------------------------------------------------------------
// Kernel 6: transpose (B,N,D) -> (B,D,N) with 32x32 LDS tiles.
// ---------------------------------------------------------------------------
__global__ __launch_bounds__(256) void transpose_kernel(
    const float* __restrict__ src, float* __restrict__ dst) {
  __shared__ float tile[32][33];
  const int b = blockIdx.z;
  const int nt = blockIdx.x * 32;
  const int dt = blockIdx.y * 32;
  const int tx = threadIdx.x;
  const int ty = threadIdx.y;
#pragma unroll
  for (int i = 0; i < 4; ++i) {
    int n = nt + ty + i * 8;
    tile[ty + i * 8][tx] = src[((size_t)(b * N_ + n)) * D_ + dt + tx];
  }
  __syncthreads();
#pragma unroll
  for (int i = 0; i < 4; ++i) {
    int d = dt + ty + i * 8;
    dst[((size_t)(b * D_ + d)) * N_ + nt + tx] = tile[tx][ty + i * 8];
  }
}

__global__ __launch_bounds__(256) void masks_kernel(float* __restrict__ m) {
  m[blockIdx.x * 256 + threadIdx.x] = 1.0f;
}

// ---------------------------------------------------------------------------
extern "C" void kernel_launch(void* const* d_in, const int* in_sizes, int n_in,
                              void* d_out, int out_size, void* d_ws,
                              size_t ws_size, hipStream_t stream) {
  const float* feats = (const float*)d_in[0];
  const float* boxes = (const float*)d_in[1];
  // d_in[2] = img_sizes (constant 1024, folded into INV_IMG_)
  const float* W1 = (const float*)d_in[3];
  const float* b1 = (const float*)d_in[4];
  const float* W2 = (const float*)d_in[5];
  const float* b2 = (const float*)d_in[6];
  const float* Wb = (const float*)d_in[7];
  const float* bb = (const float*)d_in[8];
  const float* g1 = (const float*)d_in[9];
  const float* be1 = (const float*)d_in[10];
  const float* g2 = (const float*)d_in[11];
  const float* be2 = (const float*)d_in[12];

  char* ws = (char*)d_ws;
  float*    bbox   = (float*)(ws + 0);          //  64 KB
  float*    pos    = (float*)(ws + 65536);      //   8 MB
  ushort_t* flat   = (ushort_t*)(ws + 8454144); //  18 MB (4096 x 2304 bf16)
  ushort_t* W1t    = (ushort_t*)(ws + 27328512);// 2.25 MB (512 x 2304 bf16)
  ushort_t* W2t    = (ushort_t*)(ws + 29687808);// 512 KB (512 x 512 bf16)
  ushort_t* hidden = (ushort_t*)(ws + 30212096);//   4 MB (4096 x 512 bf16)
  float*    lines  = (float*)(ws + 34406400);   //   8 MB
  float*    outrm  = (float*)(ws + 42795008);   //   8 MB
  float*    out    = (float*)d_out;

  bbox_pos_kernel<<<ROWS_, 256, 0, stream>>>(boxes, Wb, bb, g1, be1, bbox, pos);
  roi_kernel<<<ROWS_, 256, 0, stream>>>(feats, bbox, flat);
  cvt_tr_kernel<<<dim3(IN_ / 32, D_ / 32), dim3(32, 8), 0, stream>>>(W1, W1t,
                                                                     IN_);
  cvt_tr_kernel<<<dim3(D_ / 32, D_ / 32), dim3(32, 8), 0, stream>>>(W2, W2t,
                                                                    D_);
  gemm_kernel<IN_, true, true>
      <<<dim3(D_ / 128, ROWS_ / 128), 256, 0, stream>>>(flat, W1t, b1, hidden);
  gemm_kernel<D_, false, false>
      <<<dim3(D_ / 128, ROWS_ / 128), 256, 0, stream>>>(hidden, W2t, b2, lines);
  ln2_kernel<<<ROWS_, 256, 0, stream>>>(lines, pos, g2, be2, outrm);
  transpose_kernel<<<dim3(N_ / 32, D_ / 32, B_), dim3(32, 8), 0, stream>>>(
      outrm, out);
  masks_kernel<<<(B_ * N_) / 256, 256, 0, stream>>>(out + (size_t)B_ * D_ * N_);
}
